// SparseMultiHeadAttention_75282186764809
// MI455X (gfx1250) — matrix operand order repaired and checked by reading, not compiled, not run on hardware
//
#include <hip/hip_runtime.h>
#include <hip/hip_bf16.h>

// Problem constants (match reference)
#define B_  2
#define S_  2048
#define D_  1024
#define H_  16
#define HD_ 64

#define NEG_INF (-__builtin_inff())

typedef __bf16 bf16_t;
typedef bf16_t bf16x16 __attribute__((ext_vector_type(16)));
typedef float  f32x8   __attribute__((ext_vector_type(8)));

union Frag { unsigned u[8]; bf16x16 v; };

__device__ __forceinline__ unsigned short f2bf(float f) {
  unsigned x = __builtin_bit_cast(unsigned, f);
  x += 0x7FFFu + ((x >> 16) & 1u);           // round-to-nearest-even
  return (unsigned short)(x >> 16);
}

// ---------------------------------------------------------------------------
// CDNA5 async global->LDS copy (ASYNCcnt-tracked DMA into LDS, no VGPR
// round-trip).  VDST supplies the per-lane LDS byte offset; generic pointers
// to __shared__ carry the LDS offset in their low 32 bits (aperture mapping).
// ---------------------------------------------------------------------------
__device__ __forceinline__ void async_b128(void* lds, const void* g) {
  unsigned off = (unsigned)(size_t)lds;
  asm volatile("global_load_async_to_lds_b128 %0, %1, off"
               :: "v"(off), "v"(g)
               : "memory");
}
__device__ __forceinline__ void wait_async() {
  asm volatile("s_wait_asynccnt 0x0" ::: "memory");
}

// ---------------------------------------------------------------------------
// Generic GEMM: O[M,N] = A[M,K] @ W[K,N] + bias[N]
// A is f32 (projections) or bf16 (ctx, async-staged); O is bf16 or f32.
// 64x64 output tile per 256-thread block; 8 waves, each owns two 16x16 tiles.
// ---------------------------------------------------------------------------
template <bool A_BF16, bool OUT_BF16>
__global__ __launch_bounds__(256) void gemm_bias(
    const float* __restrict__ A32, const unsigned short* __restrict__ A16,
    const float* __restrict__ W, const float* __restrict__ bias,
    float* __restrict__ O32, unsigned short* __restrict__ O16,
    int M, int N, int K) {
  __shared__ __align__(16) unsigned short As[64][32];   // [m][k] bf16 bits
  __shared__ __align__(16) unsigned short Bs[64][40];   // [n][k] bf16 bits (padded)

  const int tid  = threadIdx.x;
  const int wave = tid >> 5;
  const int lane = tid & 31;
  const int m0 = blockIdx.y * 64;
  const int n0 = blockIdx.x * 64;

  const int rt  = wave & 3;           // row subtile 0..3
  const int ct0 = (wave >> 2) * 2;    // col subtiles ct0, ct0+1

  // ISA fragment lane geometry (16-bit A 16x32, B 32x16, 32-bit C/D 16x16)
  const int ml = rt * 16 + (lane & 15);
  const int kh = (lane >> 4) << 3;    // A K-half offset (0/8); also C/D row offset
  const int kb = (lane >> 4) << 4;    // 0/16; unused (B uses kh, same k order as A)

  f32x8 acc0 = {0, 0, 0, 0, 0, 0, 0, 0};
  f32x8 acc1 = {0, 0, 0, 0, 0, 0, 0, 0};

  for (int k0 = 0; k0 < K; k0 += 32) {
    if constexpr (A_BF16) {
      // pure byte copy: async DMA straight into LDS (64 rows x 64B)
      int r = tid >> 2, seg = tid & 3;
      async_b128(&As[r][seg * 8], A16 + (size_t)(m0 + r) * K + k0 + seg * 8);
    } else {
#pragma unroll
      for (int i = 0; i < 8; ++i) {                 // 64x32 = 2048 elems
        int idx = tid + i * 256;
        int r = idx >> 5, c = idx & 31;
        As[r][c] = f2bf(A32[(size_t)(m0 + r) * K + (k0 + c)]);
      }
    }
#pragma unroll
    for (int i = 0; i < 8; ++i) {                   // W tile, stored transposed
      int idx = tid + i * 256;
      int r = idx >> 6, c = idx & 63;               // r = k row, c = n col
      Bs[c][r] = f2bf(W[(size_t)(k0 + r) * N + (n0 + c)]);
    }
    if constexpr (A_BF16) wait_async();
    __syncthreads();

    Frag a;
#pragma unroll
    for (int v = 0; v < 8; ++v) {
      int kk = ((v >= 4) ? 16 : 0) + kh + 2 * (v & 3);
      a.u[v] = *reinterpret_cast<const unsigned*>(&As[ml][kk]);
    }
    Frag b0, b1;
#pragma unroll
    for (int v = 0; v < 8; ++v) {
      int kk = ((v >= 4) ? 16 : 0) + kh + 2 * (v & 3); // contiguous pair in Bs row
      b0.u[v] = *reinterpret_cast<const unsigned*>(&Bs[ct0 * 16 + (lane & 15)][kk]);
      b1.u[v] = *reinterpret_cast<const unsigned*>(&Bs[(ct0 + 1) * 16 + (lane & 15)][kk]);
    }
    acc0 = __builtin_amdgcn_wmma_f32_16x16x32_bf16(false, a.v, false, b0.v,
                                                   (short)0, acc0, false, false);
    acc1 = __builtin_amdgcn_wmma_f32_16x16x32_bf16(false, a.v, false, b1.v,
                                                   (short)0, acc1, false, false);
    __syncthreads();
  }

  const int nc0 = n0 + ct0 * 16 + (lane & 15);
  const int nc1 = nc0 + 16;
  const float bv0 = bias[nc0];
  const float bv1 = bias[nc1];
#pragma unroll
  for (int v = 0; v < 8; ++v) {
    int mg = m0 + rt * 16 + v + kh;
    float o0 = acc0[v] + bv0;
    float o1 = acc1[v] + bv1;
    if (OUT_BF16) {
      O16[(size_t)mg * N + nc0] = f2bf(o0);
      O16[(size_t)mg * N + nc1] = f2bf(o1);
    } else {
      __builtin_nontemporal_store(o0, &O32[(size_t)mg * N + nc0]);
      __builtin_nontemporal_store(o1, &O32[(size_t)mg * N + nc1]);
    }
  }
}

// ---------------------------------------------------------------------------
// QK^T 16x16 subtile over HD=64 (two K=32 WMMAs). qs: [32][64], kt: [64][64].
// ---------------------------------------------------------------------------
__device__ __forceinline__ f32x8 qk_tile(const unsigned short* qs,
                                         const unsigned short* kt,
                                         int ml, int kh, int kb, int nl) {
  f32x8 s = {0, 0, 0, 0, 0, 0, 0, 0};
#pragma unroll
  for (int st = 0; st < 2; ++st) {
    Frag a, b;
#pragma unroll
    for (int v = 0; v < 8; ++v) {
      int kk = st * 32 + ((v >= 4) ? 16 : 0) + kh + 2 * (v & 3);
      a.u[v] = *reinterpret_cast<const unsigned*>(qs + ml * 64 + kk);
      int kk2 = st * 32 + ((v >= 4) ? 16 : 0) + kh + 2 * (v & 3);        // B col = key index nl; pair contiguous
      b.u[v] = *reinterpret_cast<const unsigned*>(kt + nl * 64 + kk2);
    }
    s = __builtin_amdgcn_wmma_f32_16x16x32_bf16(false, a.v, false, b.v,
                                                (short)0, s, false, false);
  }
  return s;
}

// ---------------------------------------------------------------------------
// Fused sparse attention: one block = (b, h, 32 query rows).
// Pass 1: online row max/sum over 64-key chunks. Pass 2: recompute, normalize,
// stream attn to HBM, bf16 attn@V via WMMA into persistent accumulators.
// K/V/Q tiles staged with CDNA5 async global->LDS DMA.
// ---------------------------------------------------------------------------
__global__ __launch_bounds__(256) void sparse_attn(
    const unsigned short* __restrict__ Qb, const unsigned short* __restrict__ Kb,
    const unsigned short* __restrict__ Vb, const unsigned char* __restrict__ maskp,
    float* __restrict__ attnOut, unsigned short* __restrict__ Cb) {
  __shared__ __align__(16) unsigned short qs[32][64];
  __shared__ __align__(16) unsigned short kt[64][64];
  __shared__ __align__(16) unsigned short vt[64][64];
  __shared__ __align__(16) unsigned short at[32][64];
  __shared__ float sc[32][64];
  __shared__ float rowM[32];
  __shared__ float rowS[32];
  __shared__ float redM[32][8];
  __shared__ float redS[32][8];

  const int tid  = threadIdx.x;
  const int wave = tid >> 5;
  const int lane = tid & 31;
  const int q0 = blockIdx.x * 32;
  const int h  = blockIdx.y;
  const int b  = blockIdx.z;
  const size_t headBase = (size_t)b * S_ * D_ + (size_t)h * HD_;
  const float scale = 0.125f;   // 1/sqrt(64)

  const int rt = wave & 1;      // q-row subtile 0..1
  const int ct = wave >> 1;     // col subtile 0..3 (keys in QK^T, hd in PV)
  const int ml = rt * 16 + (lane & 15);
  const int kh = (lane >> 4) << 3;
  const int kb = (lane >> 4) << 4;
  const int nl = ct * 16 + (lane & 15);

  {                                              // Q tile 32x64 via async DMA
    int r = tid >> 3, seg = tid & 7;
    async_b128(&qs[r][seg * 8], Qb + headBase + (size_t)(q0 + r) * D_ + seg * 8);
  }
  if (tid < 32) { rowM[tid] = NEG_INF; rowS[tid] = 0.f; }
  wait_async();
  __syncthreads();

  // ---------------- Pass 1: running max / sum ----------------
  for (int kc = 0; kc < S_; kc += 64) {
#pragma unroll
    for (int i = 0; i < 2; ++i) {                // K chunk 64x64 via async DMA
      int idx = tid + i * 256;
      int r = idx >> 3, seg = idx & 7;
      async_b128(&kt[r][seg * 8], Kb + headBase + (size_t)(kc + r) * D_ + seg * 8);
    }
    wait_async();
    __syncthreads();

    f32x8 s = qk_tile(&qs[0][0], &kt[0][0], ml, kh, kb, nl);
#pragma unroll
    for (int v = 0; v < 8; ++v) {
      int m = rt * 16 + v + kh;
      int kg = kc + nl;
      float val = s[v] * scale;
      if (!maskp[(size_t)(q0 + m) * S_ + kg]) val = NEG_INF;
      sc[m][nl] = val;
    }
    __syncthreads();

    int row = tid >> 3, seg = tid & 7;
    float mx = NEG_INF;
#pragma unroll
    for (int j = 0; j < 8; ++j) mx = fmaxf(mx, sc[row][seg * 8 + j]);
    redM[row][seg] = mx;
    __syncthreads();

    float cm = NEG_INF;
#pragma unroll
    for (int j = 0; j < 8; ++j) cm = fmaxf(cm, redM[row][j]);
    float sm = 0.f;
    if (cm > NEG_INF) {
#pragma unroll
      for (int j = 0; j < 8; ++j) sm += __expf(sc[row][seg * 8 + j] - cm);
    }
    redS[row][seg] = sm;
    __syncthreads();

    if (seg == 0 && cm > NEG_INF) {
      float cs = 0.f;
#pragma unroll
      for (int j = 0; j < 8; ++j) cs += redS[row][j];
      float mo = rowM[row];
      float mn = fmaxf(mo, cm);
      float alpha = (mo > NEG_INF) ? __expf(mo - mn) : 0.f;
      float beta  = __expf(cm - mn);
      rowS[row] = rowS[row] * alpha + cs * beta;
      rowM[row] = mn;
    }
    __syncthreads();
  }

  if (tid < 32) rowS[tid] = 1.f / rowS[tid];     // store inverse sum
  __syncthreads();

  // ---------------- Pass 2: normalize + write attn + PV ----------------
  f32x8 acc = {0, 0, 0, 0, 0, 0, 0, 0};
  for (int kc = 0; kc < S_; kc += 64) {
#pragma unroll
    for (int i = 0; i < 2; ++i) {                // K and V chunks via async DMA
      int idx = tid + i * 256;
      int r = idx >> 3, seg = idx & 7;
      size_t src = headBase + (size_t)(kc + r) * D_ + seg * 8;
      async_b128(&kt[r][seg * 8], Kb + src);
      async_b128(&vt[r][seg * 8], Vb + src);
    }
    wait_async();
    __syncthreads();

    f32x8 s = qk_tile(&qs[0][0], &kt[0][0], ml, kh, kb, nl);
#pragma unroll
    for (int v = 0; v < 8; ++v) {
      int m = rt * 16 + v + kh;
      int kg = kc + nl;
      float val = s[v] * scale;
      if (!maskp[(size_t)(q0 + m) * S_ + kg]) val = NEG_INF;
      float p = __expf(val - rowM[m]) * rowS[m];
      size_t aidx = ((size_t)(b * H_ + h) * S_ + (q0 + m)) * S_ + kg;
      __builtin_nontemporal_store(p, &attnOut[aidx]);   // stream; keep L2 for K/V/mask
      at[m][nl] = f2bf(p);
    }
    __syncthreads();

    // attn (32 x 64-chunk) @ V (64-chunk x 64) -> accumulate ctx subtile
#pragma unroll
    for (int st = 0; st < 2; ++st) {
      Frag a, bv;
#pragma unroll
      for (int v = 0; v < 8; ++v) {
        int kk = st * 32 + ((v >= 4) ? 16 : 0) + kh + 2 * (v & 3);
        a.u[v] = *reinterpret_cast<const unsigned*>(&at[ml][kk]);
        int kk2 = st * 32 + ((v >= 4) ? 16 : 0) + kh + 2 * (v & 3);
        unsigned lo = vt[kk2][nl];
        unsigned hi = vt[kk2 + 1][nl];
        bv.u[v] = lo | (hi << 16);
      }
      acc = __builtin_amdgcn_wmma_f32_16x16x32_bf16(false, a.v, false, bv.v,
                                                    (short)0, acc, false, false);
    }
    __syncthreads();
  }

#pragma unroll
  for (int v = 0; v < 8; ++v) {
    int m = rt * 16 + v + kh;
    Cb[headBase + (size_t)(q0 + m) * D_ + nl] = f2bf(acc[v]);
  }
}

// ---------------------------------------------------------------------------
extern "C" void kernel_launch(void* const* d_in, const int* in_sizes, int n_in,
                              void* d_out, int out_size, void* d_ws, size_t ws_size,
                              hipStream_t stream) {
  (void)in_sizes; (void)n_in; (void)out_size; (void)ws_size;
  const float* query = (const float*)d_in[0];
  const float* key_  = (const float*)d_in[1];
  const float* value = (const float*)d_in[2];
  const float* Wq = (const float*)d_in[3];
  const float* bq = (const float*)d_in[4];
  const float* Wk = (const float*)d_in[5];
  const float* bk = (const float*)d_in[6];
  const float* Wv = (const float*)d_in[7];
  const float* bv = (const float*)d_in[8];
  const float* Wo = (const float*)d_in[9];
  const float* bo = (const float*)d_in[10];
  const unsigned char* maskp = (const unsigned char*)d_in[11];

  const size_t nQKV = (size_t)B_ * S_ * D_;      // 4 Mi elems
  unsigned short* Qb = (unsigned short*)d_ws;    // bf16 Q   (8 MB)
  unsigned short* Kb = Qb + nQKV;                // bf16 K   (8 MB)
  unsigned short* Vb = Kb + nQKV;                // bf16 V   (8 MB)
  unsigned short* Cb = Vb + nQKV;                // bf16 ctx (8 MB)

  float* out  = (float*)d_out;                   // [B,S,D]
  float* attn = out + nQKV;                      // [B,H,S,S]

  const int M = B_ * S_, N = D_, K = D_;
  dim3 gGemm(N / 64, M / 64);                    // (16, 64)
  dim3 blk(256);

  gemm_bias<false, true><<<gGemm, blk, 0, stream>>>(query, nullptr, Wq, bq,
                                                    nullptr, Qb, M, N, K);
  gemm_bias<false, true><<<gGemm, blk, 0, stream>>>(key_, nullptr, Wk, bk,
                                                    nullptr, Kb, M, N, K);
  gemm_bias<false, true><<<gGemm, blk, 0, stream>>>(value, nullptr, Wv, bv,
                                                    nullptr, Vb, M, N, K);

  dim3 gAttn(S_ / 32, H_, B_);                   // (64, 16, 2)
  sparse_attn<<<gAttn, blk, 0, stream>>>(Qb, Kb, Vb, maskp, attn, Cb);

  gemm_bias<true, false><<<gGemm, blk, 0, stream>>>(nullptr, Cb, Wo, bo,
                                                    out, nullptr, M, N, K);
}
